// Dense_Net_OpticalFront_7267084664998
// MI455X (gfx1250) — matrix operand order repaired and checked by reading, not compiled, not run on hardware
//
#include <hip/hip_runtime.h>
#include <hip/hip_bf16.h>
#include <math.h>

#define HDIM 28
#define HW   784
#define HID  800
#define NBAT 32768
#define KP   800                 // K padded 784 -> 800 (25 x 32)
#define TWO_PI_28 0.2243994752564138f

typedef __attribute__((ext_vector_type(16))) __bf16 v16bf;
typedef __attribute__((ext_vector_type(8)))  float  v8f;

union Frag { v16bf v; uint4 q[2]; };

__device__ __forceinline__ unsigned short f2bf(float f) {
  unsigned int u = __float_as_uint(f);
  u += 0x7fffu + ((u >> 16) & 1u);     // round-to-nearest-even
  return (unsigned short)(u >> 16);
}

// ---------------- setup: fold S F^-1 S diag(m) S F S into w1 ----------------
// Dinv(u,k) = (1/28) e^{+i 2pi ((u+14)(k-14) mod 28)/28}
// Dfwd(k,n) =        e^{-i 2pi ((k+14)(n-14) mod 28)/28}

__global__ void __launch_bounds__(256) opt_s1(const float* __restrict__ w1,
                                              float2* __restrict__ P) {
  int idx = blockIdx.x * 256 + threadIdx.x;
  if (idx >= HID * HW) return;
  int h = idx / HW, kv = idx % HW, k = kv / HDIM, v = kv % HDIM;
  float re = 0.f, im = 0.f;
  for (int u = 0; u < HDIM; ++u) {
    int e = ((u + 14) * (k - 14)) % HDIM; if (e < 0) e += HDIM;
    float s, c; __sincosf(TWO_PI_28 * (float)e, &s, &c);
    float w = w1[h * HW + u * HDIM + v];
    re += w * c; im += w * s;
  }
  P[idx] = make_float2(re * (1.f / 28.f), im * (1.f / 28.f));
}

__global__ void __launch_bounds__(256) opt_s2(const float2* __restrict__ P,
                                              const float* __restrict__ mre,
                                              const float* __restrict__ mim,
                                              float2* __restrict__ T) {
  int idx = blockIdx.x * 256 + threadIdx.x;
  if (idx >= HID * HW) return;
  int h = idx / HW, kl = idx % HW, k = kl / HDIM, l = kl % HDIM;
  int base = (h * HDIM + k) * HDIM;
  float re = 0.f, im = 0.f;
  for (int v = 0; v < HDIM; ++v) {
    int e = ((v + 14) * (l - 14)) % HDIM; if (e < 0) e += HDIM;
    float s, c; __sincosf(TWO_PI_28 * (float)e, &s, &c);
    float2 p = P[base + v];
    re += p.x * c - p.y * s;
    im += p.x * s + p.y * c;
  }
  re *= (1.f / 28.f); im *= (1.f / 28.f);
  float a = mre[kl], b = mim[kl];
  T[idx] = make_float2(re * a - im * b, re * b + im * a);
}

__global__ void __launch_bounds__(256) opt_s3(const float2* __restrict__ T,
                                              float2* __restrict__ Q) {
  int idx = blockIdx.x * 256 + threadIdx.x;
  if (idx >= HID * HW) return;
  int h = idx / HW, kt = idx % HW, k = kt / HDIM, t = kt % HDIM;
  int base = (h * HDIM + k) * HDIM;
  float re = 0.f, im = 0.f;
  for (int l = 0; l < HDIM; ++l) {
    int e = ((l + 14) * (t - 14)) % HDIM; if (e < 0) e += HDIM;
    float s, c; __sincosf(TWO_PI_28 * (float)e, &s, &c);
    float2 q = T[base + l];                 // * (c - i s)
    re += q.x * c + q.y * s;
    im += q.y * c - q.x * s;
  }
  Q[idx] = make_float2(re, im);
}

__global__ void __launch_bounds__(256) opt_s4(const float2* __restrict__ Q,
                                              unsigned short* __restrict__ W1b) {
  int idx = blockIdx.x * 256 + threadIdx.x;
  if (idx >= HID * KP) return;
  int h = idx / KP, kp = idx % KP;
  if (kp >= HW) { W1b[idx] = 0; return; }   // zero pad K 784..799
  int ss = kp / HDIM, tt = kp % HDIM;
  float re = 0.f;
  for (int k = 0; k < HDIM; ++k) {
    int e = ((k + 14) * (ss - 14)) % HDIM; if (e < 0) e += HDIM;
    float s, c; __sincosf(TWO_PI_28 * (float)e, &s, &c);
    float2 q = Q[(h * HDIM + k) * HDIM + tt];   // Re((qx+iqy)(c-is))
    re += q.x * c + q.y * s;
  }
  W1b[idx] = f2bf(re);
}

__global__ void __launch_bounds__(256) opt_s5(const float* __restrict__ w4,
                                              unsigned short* __restrict__ w4b) {
  int idx = blockIdx.x * 256 + threadIdx.x;
  if (idx >= 16 * HID) return;
  int n = idx / HID, k = idx % HID;
  w4b[idx] = (n < 10) ? f2bf(w4[n * HID + k]) : (unsigned short)0;
}

__global__ void __launch_bounds__(256) opt_s6(const float* __restrict__ x,
                                              unsigned short* __restrict__ Xb) {
  int idx = blockIdx.x * 256 + threadIdx.x;
  if (idx >= NBAT * KP) return;
  int r = idx / KP, k = idx % KP;
  Xb[idx] = (k < HW) ? f2bf(x[r * HW + k]) : (unsigned short)0;
}

// ---------------- fused MLP: 64 rows/block, bf16 WMMA, fp32 accum ----------------
__global__ void __launch_bounds__(256, 1) fused_mlp(
    const unsigned short* __restrict__ Xb,    // [NBAT][KP] bf16
    const unsigned short* __restrict__ W1b,   // [HID][KP]  bf16
    const unsigned short* __restrict__ w4b,   // [16][HID]  bf16
    const float* __restrict__ b1,
    const float* __restrict__ b4,
    float* __restrict__ out) {
  extern __shared__ char smem[];
  unsigned short* sx = (unsigned short*)smem;             // [64][KP] 100 KB
  unsigned short* sh = (unsigned short*)(smem + 102400);  // [64][KP] 100 KB
  float* sl = (float*)smem;                               // [64][16] reused post-L1

  const int tid  = threadIdx.x;
  const int row0 = blockIdx.x * 64;

  { // stage X tile: async global->LDS DMA (bypasses VGPRs, tracked by ASYNCcnt)
    const unsigned lds_base = (unsigned)(size_t)(void*)sx;  // flat addr low 32b = LDS offset
    const unsigned long long gbase =
        (unsigned long long)(const void*)(Xb + (size_t)row0 * KP);
    for (int i = tid; i < 64 * KP / 8; i += 256) {
      unsigned           ldsa = lds_base + (unsigned)i * 16u;
      unsigned long long ga   = gbase + (unsigned long long)i * 16u;
      asm volatile("global_load_async_to_lds_b128 %0, %1, off"
                   :: "v"(ldsa), "v"(ga) : "memory");
    }
    asm volatile("s_wait_asynccnt 0" ::: "memory");
  }
  __syncthreads();

  const int lane = tid & 31, wid = tid >> 5;
  const int l16 = lane & 15, hi = lane >> 4;
  const int m0 = (wid & 3) * 16, nh = wid >> 2;

  // ---- layer 1: h = relu(X @ W1eff^T + b1), N in 5 chunks of 160 ----
  for (int c = 0; c < 5; ++c) {
    v8f acc[5] = {};
    const int nbase = c * 160 + nh * 80;
    const unsigned short* arow = sx + (m0 + l16) * KP;
    for (int ks = 0; ks < 25; ++ks) {
      const int k0 = ks * 32;
      Frag a;                                       // A: 16 contiguous K per lane (K 0..15 / 16..31)
      a.q[0] = *(const uint4*)(arow + k0 + 16 * hi);
      a.q[1] = *(const uint4*)(arow + k0 + 16 * hi + 8);
#pragma unroll
      for (int j = 0; j < 5; ++j) {
        const unsigned short* brow =
            W1b + (size_t)(nbase + j * 16 + l16) * KP + k0 + 16 * hi;
        Frag b;                                     // B: 16 contiguous K per lane
        b.q[0] = *(const uint4*)(brow);
        b.q[1] = *(const uint4*)(brow + 8);
        acc[j] = __builtin_amdgcn_wmma_f32_16x16x32_bf16(
            false, a.v, false, b.v, (short)0, acc[j], false, false);
      }
    }
#pragma unroll
    for (int j = 0; j < 5; ++j) {
      const int n = nbase + j * 16 + l16;
      const float bv = b1[n];
#pragma unroll
      for (int r = 0; r < 8; ++r) {                 // C: lanes0-15 M=r, lanes16-31 M=8+r
        float v = acc[j][r] + bv;
        v = v > 0.f ? v : 0.f;
        sh[(m0 + r + 8 * hi) * KP + n] = f2bf(v);
      }
    }
  }
  __syncthreads();

  // ---- layer 2: logits = h @ w4^T + b4 (N padded 10->16), waves 0..3 ----
  if (wid < 4) {
    const int mm0 = wid * 16;
    v8f acc2 = {};
    const unsigned short* arow = sh + (mm0 + l16) * KP;
    for (int ks = 0; ks < 25; ++ks) {
      const int k0 = ks * 32;
      Frag a, b;
      a.q[0] = *(const uint4*)(arow + k0 + 16 * hi);
      a.q[1] = *(const uint4*)(arow + k0 + 16 * hi + 8);
      const unsigned short* brow = w4b + l16 * HID + k0 + 16 * hi;
      b.q[0] = *(const uint4*)(brow);
      b.q[1] = *(const uint4*)(brow + 8);
      acc2 = __builtin_amdgcn_wmma_f32_16x16x32_bf16(
          false, a.v, false, b.v, (short)0, acc2, false, false);
    }
    const float bb = (l16 < 10) ? b4[l16] : 0.f;
#pragma unroll
    for (int r = 0; r < 8; ++r)
      sl[(mm0 + r + 8 * hi) * 16 + l16] = acc2[r] + bb;
  }
  __syncthreads();

  // ---- log-softmax over 10, one thread per row ----
  if (tid < 64) {
    float v[10], mx = -1e30f;
#pragma unroll
    for (int j = 0; j < 10; ++j) { v[j] = sl[tid * 16 + j]; mx = fmaxf(mx, v[j]); }
    float sum = 0.f;
#pragma unroll
    for (int j = 0; j < 10; ++j) sum += __expf(v[j] - mx);
    const float ls = __logf(sum);
    float* o = out + (size_t)(row0 + tid) * 10;
#pragma unroll
    for (int j = 0; j < 10; ++j) o[j] = v[j] - mx - ls;
  }
}

extern "C" void kernel_launch(void* const* d_in, const int* in_sizes, int n_in,
                              void* d_out, int out_size, void* d_ws, size_t ws_size,
                              hipStream_t stream) {
  (void)in_sizes; (void)n_in; (void)out_size; (void)ws_size;
  const float* x   = (const float*)d_in[0];
  const float* tre = (const float*)d_in[1];
  const float* tim = (const float*)d_in[2];
  const float* w1  = (const float*)d_in[3];
  const float* b1  = (const float*)d_in[4];
  const float* w4  = (const float*)d_in[5];
  const float* b4  = (const float*)d_in[6];
  float* out = (float*)d_out;

  char* ws = (char*)d_ws;
  float2*         C1  = (float2*)(ws);                       // 800*784 cplx  5.02 MB
  float2*         C2  = (float2*)(ws + 5017600);             // 5.02 MB
  unsigned short* W1b = (unsigned short*)(ws + 10035200);    // 800*800 bf16  1.28 MB
  unsigned short* w4b = (unsigned short*)(ws + 11315200);    // 16*800 bf16
  unsigned short* Xb  = (unsigned short*)(ws + 11340800);    // 32768*800 bf16 52.4 MB

  opt_s1<<<2450, 256, 0, stream>>>(w1, C1);
  opt_s2<<<2450, 256, 0, stream>>>(C1, tre, tim, C2);
  opt_s3<<<2450, 256, 0, stream>>>(C2, C1);
  opt_s4<<<2500, 256, 0, stream>>>(C1, W1b);
  opt_s5<<<50,   256, 0, stream>>>(w4, w4b);
  opt_s6<<<102400, 256, 0, stream>>>(x, Xb);

  hipFuncSetAttribute(reinterpret_cast<const void*>(fused_mlp),
                      hipFuncAttributeMaxDynamicSharedMemorySize, 204800);
  fused_mlp<<<512, 256, 204800, stream>>>(Xb, W1b, w4b, b1, b4, out);
}
